// FeaturePropagation_75084618268926
// MI455X (gfx1250) — matrix operand order repaired and checked by reading, not compiled, not run on hardware
//
#include <hip/hip_runtime.h>

// ---------- types ----------
typedef __bf16 v16bf __attribute__((ext_vector_type(16)));
typedef float  v8f   __attribute__((ext_vector_type(8)));
typedef int    i4    __attribute__((ext_vector_type(4)));
typedef unsigned int u32;
typedef unsigned short u16;

union FragAB { i4 q[2]; v16bf bf; };

// scheduling barrier (keep load group ahead of WMMA group)
#if defined(__has_builtin)
#  if __has_builtin(__builtin_amdgcn_sched_barrier)
#    define SCHED_BARRIER() __builtin_amdgcn_sched_barrier(0)
#  endif
#endif
#ifndef SCHED_BARRIER
#  define SCHED_BARRIER() ((void)0)
#endif

// ---------- constants ----------
#define BATCH 8
#define N1    4096
#define N2    1024
#define F1    128
#define F2    256
#define KDIM  384          // F2 + F1
#define HDIM  256
#define MTOT  (BATCH * N1) // 32768
#define LDH   264          // padded LDS stride (bf16 elems) for H tile
#define LEAKY 0.2f

__device__ __forceinline__ u16 f2bf(float f) {
    u32 u = __float_as_uint(f);
    u32 r = (u + 0x7FFFu + ((u >> 16) & 1u)) >> 16;   // round-to-nearest-even
    return (u16)r;
}

__device__ __forceinline__ float leaky(float x) { return x >= 0.0f ? x : LEAKY * x; }

__device__ __forceinline__ FragAB loadFrag(const u32* p) {
    FragAB f;
    f.q[0] = *(const i4*)p;
    f.q[1] = *(const i4*)(p + 4);
    return f;
}

// =====================================================================
// Kernel 1: pack W (KxN f32, row-major) into WMMA B-fragment dword order.
// B 32x16 bf16 order used here (contiguous k, as for A): lane n (0-15) = column; VGPR r: lanes0-15 K=2r,2r+1;
// lanes16-31 K=16+2r,17+2r.  packed[chunk*256 + lane*8 + r], chunk = kt*(N/16)+nt.
// =====================================================================
__global__ void pack_w_kernel(const float* __restrict__ W, u32* __restrict__ out,
                              int K, int N) {
    int t = blockIdx.x * 256 + threadIdx.x;
    int total = (K * N) >> 1;
    if (t >= total) return;
    int r     = t & 7;
    int lane  = (t >> 3) & 31;
    int chunk = t >> 8;
    int ntiles = N >> 4;
    int kt = chunk / ntiles;
    int nt = chunk - kt * ntiles;
    int half = lane >> 4;
    int n = nt * 16 + (lane & 15);
    int k = kt * 32 + 2 * r + 16 * half;
    u32 lo = f2bf(W[(size_t)k * N + n]);
    u32 hi = f2bf(W[(size_t)(k + 1) * N + n]);
    out[t] = lo | (hi << 16);
}

// =====================================================================
// Kernel 2: 3-NN inverse-distance interpolation + concat -> X bf16 (MTOT x 384)
// One wave per query point; xyz2 tile cached in LDS; top-3 via shfl_xor merge.
// =====================================================================
__device__ __forceinline__ void ins3(float d, int j,
                                     float& d0, float& d1, float& d2,
                                     int& i0, int& i1, int& i2) {
    if (d < d2) {
        d2 = d; i2 = j;
        if (d2 < d1) { float td = d1; d1 = d2; d2 = td; int ti = i1; i1 = i2; i2 = ti; }
        if (d1 < d0) { float td = d0; d0 = d1; d1 = td; int ti = i0; i0 = i1; i1 = ti; }
    }
}

__global__ void interp_kernel(const float* __restrict__ xyz1,
                              const float* __restrict__ xyz2,
                              const float* __restrict__ points1,
                              const float* __restrict__ points2,
                              u16* __restrict__ X) {
    __shared__ float sxyz[N2 * 3];                    // 12 KB
    const int b     = blockIdx.x >> 4;                // 16 blocks per batch
    const int chunk = blockIdx.x & 15;
    const int tid   = threadIdx.x;
    const int wave  = tid >> 5;
    const int lane  = tid & 31;

    for (int i = tid; i < N2 * 3; i += 256)
        sxyz[i] = xyz2[(size_t)b * (N2 * 3) + i];
    __syncthreads();

    const float* p2base = points2 + (size_t)b * N2 * F2;

    for (int it = 0; it < 32; ++it) {
        const int n1 = chunk * 256 + wave * 32 + it;
        const size_t row = (size_t)b * N1 + n1;
        const float* p = xyz1 + row * 3;
        const float px = p[0], py = p[1], pz = p[2];

        float d0 = 3.0e38f, d1 = 3.0e38f, d2 = 3.0e38f;
        int   i0 = 0, i1 = 0, i2 = 0;
        for (int j = lane; j < N2; j += 32) {
            float dx = px - sxyz[j * 3 + 0];
            float dy = py - sxyz[j * 3 + 1];
            float dz = pz - sxyz[j * 3 + 2];
            float d = dx * dx + dy * dy + dz * dz;
            ins3(d, j, d0, d1, d2, i0, i1, i2);
        }
        // wave32 butterfly merge of per-lane top-3
        for (int off = 16; off > 0; off >>= 1) {
            float e0 = __shfl_xor(d0, off, 32); int j0 = __shfl_xor(i0, off, 32);
            float e1 = __shfl_xor(d1, off, 32); int j1 = __shfl_xor(i1, off, 32);
            float e2 = __shfl_xor(d2, off, 32); int j2 = __shfl_xor(i2, off, 32);
            ins3(e0, j0, d0, d1, d2, i0, i1, i2);
            ins3(e1, j1, d0, d1, d2, i0, i1, i2);
            ins3(e2, j2, d0, d1, d2, i0, i1, i2);
        }
        float w0 = 1.0f / fmaxf(d0, 1e-10f);
        float w1 = 1.0f / fmaxf(d1, 1e-10f);
        float w2 = 1.0f / fmaxf(d2, 1e-10f);
        float s = 1.0f / (w0 + w1 + w2);
        w0 *= s; w1 *= s; w2 *= s;

        const float* a0 = p2base + (size_t)i0 * F2;
        const float* a1 = p2base + (size_t)i1 * F2;
        const float* a2 = p2base + (size_t)i2 * F2;
        u16* xr = X + row * KDIM;
        #pragma unroll
        for (int t2 = 0; t2 < 8; ++t2) {
            int f = lane + 32 * t2;
            xr[f] = f2bf(w0 * a0[f] + w1 * a1[f] + w2 * a2[f]);
        }
        const float* p1 = points1 + row * F1;
        #pragma unroll
        for (int t2 = 0; t2 < 4; ++t2) {
            int f = lane + 32 * t2;
            xr[F2 + f] = f2bf(p1[f]);
        }
    }
}

// =====================================================================
// Kernel 3: fused MLP.  Block = 256 thr = 8 waves, 64x256 output tile.
// Wave (wr,wc): rows wr*16..+16, cols wc*128..+128 (eight 16x16 accum tiles).
// Stage1: X(bf16) x pw1 -> H tile (bias+leaky, bf16) in LDS.
// Stage2: H x pw2 -> out (bias+leaky, f32).
// Fully unrolled K-loops: per K-step, an 18-load clause (8 B frags + A frag, 2 x 16 B each),
// sched_barrier, then the 8-WMMA chain.  Stable accumulator allocation
// (no loop-carried register rotation); loads of step kt+1 overlap WMMAs
// of step kt.
// =====================================================================
__global__ void __launch_bounds__(256, 1)
fp_mlp_kernel(const u32* __restrict__ xw,     // X as dwords (row stride 192)
              const u32* __restrict__ pw1,
              const u32* __restrict__ pw2,
              const float* __restrict__ bias1,
              const float* __restrict__ bias2,
              float* __restrict__ out) {
    __shared__ __align__(16) u16 hbuf[64 * LDH];      // 33.8 KB

    const int tid  = threadIdx.x;
    const int lane = tid & 31;
    const int w    = tid >> 5;
    const int wr   = w & 3;           // 4 wave-rows
    const int wc   = w >> 2;          // 2 wave-cols
    const int half = lane >> 4;
    const int l15  = lane & 15;
    const int row0 = blockIdx.x * 64;

    v8f c[8] = {};

    // ---------------- Stage 1: K = 384, 12 steps ----------------
    const int arow = row0 + wr * 16 + l15;
    const u32* abase = xw + (size_t)arow * 192 + half * 8;
    const u32* b1base = pw1 + (size_t)(wc * 8) * 256 + lane * 8;

    #pragma unroll
    for (int kt = 0; kt < 12; ++kt) {
        FragAB b[8];
        #pragma unroll
        for (int j = 0; j < 8; ++j)
            b[j] = loadFrag(b1base + ((size_t)kt * 16 + j) * 256);
        FragAB a = loadFrag(abase + kt * 16);
        SCHED_BARRIER();                               // loads stay ahead of WMMAs
        #pragma unroll
        for (int j = 0; j < 8; ++j)
            c[j] = __builtin_amdgcn_wmma_f32_16x16x32_bf16(
                false, a.bf, false, b[j].bf, (short)0, c[j], false, false);
    }

    // epilogue 1: bias + leaky -> bf16 into LDS H tile
    #pragma unroll
    for (int j = 0; j < 8; ++j) {
        const int n = wc * 128 + j * 16 + l15;
        const float bv = bias1[n];
        #pragma unroll
        for (int v = 0; v < 8; ++v) {
            const int m = wr * 16 + v + 8 * half;
            hbuf[m * LDH + n] = f2bf(leaky(c[j][v] + bv));
        }
        c[j] = (v8f)0.0f;
    }
    __syncthreads();

    // ---------------- Stage 2: K = 256, 8 steps ----------------
    const u32* hb = ((const u32*)&hbuf[(wr * 16 + l15) * LDH]) + half * 8;
    const u32* b2base = pw2 + (size_t)(wc * 8) * 256 + lane * 8;

    #pragma unroll
    for (int kt = 0; kt < 8; ++kt) {
        FragAB b[8];
        #pragma unroll
        for (int j = 0; j < 8; ++j)
            b[j] = loadFrag(b2base + ((size_t)kt * 16 + j) * 256);
        FragAB a2;
        a2.q[0] = *(const i4*)(hb + kt * 16);
        a2.q[1] = *(const i4*)(hb + kt * 16 + 4);
        SCHED_BARRIER();
        #pragma unroll
        for (int j = 0; j < 8; ++j)
            c[j] = __builtin_amdgcn_wmma_f32_16x16x32_bf16(
                false, a2.bf, false, b[j].bf, (short)0, c[j], false, false);
    }

    // epilogue 2: bias + leaky -> f32 global
    #pragma unroll
    for (int j = 0; j < 8; ++j) {
        const int n = wc * 128 + j * 16 + l15;
        const float bv = bias2[n];
        #pragma unroll
        for (int v = 0; v < 8; ++v) {
            const size_t m = (size_t)row0 + wr * 16 + v + 8 * half;
            out[m * HDIM + n] = leaky(c[j][v] + bv);
        }
    }
}

// =====================================================================
extern "C" void kernel_launch(void* const* d_in, const int* in_sizes, int n_in,
                              void* d_out, int out_size, void* d_ws, size_t ws_size,
                              hipStream_t stream) {
    const float* xyz1    = (const float*)d_in[0];
    const float* xyz2    = (const float*)d_in[1];
    const float* points1 = (const float*)d_in[2];
    const float* points2 = (const float*)d_in[3];
    const float* W1      = (const float*)d_in[4];
    const float* b1      = (const float*)d_in[5];
    const float* W2      = (const float*)d_in[6];
    const float* b2      = (const float*)d_in[7];
    float* out = (float*)d_out;

    // workspace layout
    const size_t xBytes   = (size_t)MTOT * KDIM * 2;          // 25,165,824
    const size_t pw1Bytes = (size_t)(KDIM / 32) * (HDIM / 16) * 256 * 4; // 196,608
    u16* X   = (u16*)d_ws;
    u32* pw1 = (u32*)((char*)d_ws + xBytes);
    u32* pw2 = (u32*)((char*)d_ws + xBytes + pw1Bytes);

    // pack weights into WMMA B-fragment order (bf16 pairs per dword)
    {
        int total1 = (KDIM * HDIM) >> 1;                      // 49152
        pack_w_kernel<<<(total1 + 255) / 256, 256, 0, stream>>>(W1, pw1, KDIM, HDIM);
        int total2 = (HDIM * HDIM) >> 1;                      // 32768
        pack_w_kernel<<<(total2 + 255) / 256, 256, 0, stream>>>(W2, pw2, HDIM, HDIM);
    }

    // 3-NN interpolation + concat -> X (bf16)
    interp_kernel<<<BATCH * (N1 / 256), 256, 0, stream>>>(xyz1, xyz2, points1, points2, X);

    // fused 2-layer MLP with WMMA
    fp_mlp_kernel<<<MTOT / 64, 256, 0, stream>>>((const u32*)X, pw1, pw2, b1, b2, out);
}
